// VQVAE_34265249088231
// MI455X (gfx1250) — matrix operand order repaired and checked by reading, not compiled, not run on hardware
//
#include <hip/hip_runtime.h>
#include <hip/hip_bf16.h>

// ---------------------------------------------------------------------------
// VQ-VAE pipeline for MI455X (gfx1250, wave32, WMMA).
// bf16 WMMA GEMMs for fc11/fc12 + all 8 LSTM input projections.
// GEMM staging uses gfx1250 async global->LDS copies (ASYNCcnt) with LDS
// double buffering (compile-time buffer ids, branch-free pipeline) so the
// DMA of K-tile i+1 overlaps the WMMAs of tile i.
// Sequential LSTM recurrences with float4-vectorized Whh matvec.
// ---------------------------------------------------------------------------

typedef __attribute__((ext_vector_type(16))) __bf16 v16bf;
typedef __attribute__((ext_vector_type(8)))  float  v8f;

#define T_STEPS 4096
#define HID     511
#define G4      2044          // 4*HID
#define G4P     2048          // padded gate rows
#define LDA     1024          // padded row stride for bf16 activations
#define LDP     2048          // padded row stride for f32 gate pre-activations
#define HSTR    512           // row stride for per-direction hidden outputs

// ---------------------------------------------------------------------------
// WMMA GEMM: C[4096, N] = act(Abf[4096, Kp] x Wbf[Np, Kp]^T + bias)
// Abf: bf16, row stride lda, zero-padded. Wbf: bf16, padded [Np][Kp], zeroed.
// Requires Kp % 64 == 0 (two 32-wide K-tiles per unrolled iteration).
// Block = 128 threads (4 waves), tile 64(M) x 64(N).
// Each wave: 1 A-fragment reused across 4 N-subtiles -> 4 v_wmma per K-step.
// ---------------------------------------------------------------------------
__global__ __launch_bounds__(128)
void k_gemm_wmma(const __bf16* __restrict__ A, int lda,
                 const __bf16* __restrict__ W, int Kp,
                 const float* __restrict__ bias,
                 void* __restrict__ Cout, int ldc, int N,
                 int relu, int out_bf16)
{
    __shared__ __align__(16) __bf16 sA[2][64 * 32];   // [m][k] double-buffered
    __shared__ __align__(16) __bf16 sW[2][64 * 32];   // [n][k] double-buffered

    const int tid   = threadIdx.x;
    const int lane  = tid & 31;
    const int wv    = tid >> 5;               // 0..3 -> 16-row M sub-tile
    const int tileN = blockIdx.x * 64;
    const int rowB  = blockIdx.y * 64;

    // 256 chunks of 8 bf16 (16B) per 64x32 tile; 2 chunks per thread
    const int c0 = tid, c1 = tid + 128;
    const int rA0 = c0 >> 2, kA0 = (c0 & 3) * 8;
    const int rA1 = c1 >> 2, kA1 = (c1 & 3) * 8;

    // gfx1250 async global->LDS 16B copy, tracked by ASYNCcnt
    auto async_cp = [&](const __bf16* g, __bf16* l) {
        unsigned lds_off = (unsigned)(unsigned long long)l;   // low 32b = LDS offset
        unsigned long long ga = (unsigned long long)g;
        asm volatile("global_load_async_to_lds_b128 %0, %1, off"
                     :: "v"(lds_off), "v"(ga) : "memory");
    };

    v8f acc0 = {}, acc1 = {}, acc2 = {}, acc3 = {};

    #define STAGE(BUF, K0)                                                        \
        do {                                                                      \
            async_cp(A + (size_t)(rowB  + rA0) * lda + (K0) + kA0, &sA[BUF][c0 * 8]); \
            async_cp(A + (size_t)(rowB  + rA1) * lda + (K0) + kA1, &sA[BUF][c1 * 8]); \
            async_cp(W + (size_t)(tileN + rA0) * Kp  + (K0) + kA0, &sW[BUF][c0 * 8]); \
            async_cp(W + (size_t)(tileN + rA1) * Kp  + (K0) + kA1, &sW[BUF][c1 * 8]); \
        } while (0)

    #define WAIT_SWAP()                                          \
        do {                                                     \
            asm volatile("s_wait_asynccnt 0x0" ::: "memory");    \
            __syncthreads();                                     \
        } while (0)

    #define COMPUTE(BUF)                                                          \
        do {                                                                      \
            v16bf af;                                                             \
            {                                                                     \
                const int m  = lane & 15;                                         \
                const int kb = (lane >> 4) * 8;                                   \
                _Pragma("unroll")                                                 \
                for (int e = 0; e < 16; ++e) {                                    \
                    int kk = (e < 8) ? (kb + e) : (16 + kb + (e - 8));            \
                    af[e] = sA[BUF][(wv * 16 + m) * 32 + kk];                     \
                }                                                                 \
            }                                                                     \
            const int nl  = lane & 15;                                            \
            const int kb2 = (lane >> 4) * 8;                                      \
            _Pragma("unroll")                                                     \
            for (int nt = 0; nt < 4; ++nt) {                                      \
                v16bf bf;                                                         \
                _Pragma("unroll")                                                 \
                for (int e = 0; e < 16; ++e)                                      \
                    bf[e] = sW[BUF][(nt * 16 + nl) * 32 + ((e < 8) ? (kb2 + e) : (16 + kb2 + (e - 8)))]; \
                v8f& acc = (nt == 0) ? acc0 : (nt == 1) ? acc1                    \
                         : (nt == 2) ? acc2 : acc3;                               \
                acc = __builtin_amdgcn_wmma_f32_16x16x32_bf16(                    \
                          false, af, false, bf, (short)0, acc, false, false);     \
            }                                                                     \
        } while (0)

    // branch-free double-buffered pipeline (Kp % 64 == 0)
    STAGE(0, 0);
    WAIT_SWAP();
    for (int k0 = 0; k0 < Kp; k0 += 64) {
        STAGE(1, k0 + 32);                   // always in range: Kp % 64 == 0
        COMPUTE(0);
        WAIT_SWAP();
        const int kn = (k0 + 64 < Kp) ? (k0 + 64) : 0;   // last stage = dummy
        STAGE(0, kn);
        COMPUTE(1);
        WAIT_SWAP();
    }
    #undef STAGE
    #undef WAIT_SWAP
    #undef COMPUTE

    // C layout: lane n=lane&15; vgpr r -> row (lane>>4)*8 + r of the 16-row tile
    #pragma unroll
    for (int nt = 0; nt < 4; ++nt) {
        const v8f& acc = (nt == 0) ? acc0 : (nt == 1) ? acc1 : (nt == 2) ? acc2 : acc3;
        int n = tileN + nt * 16 + (lane & 15);
        if (n < N) {
            float b = bias ? bias[n] : 0.f;
            #pragma unroll
            for (int r = 0; r < 8; ++r) {
                int mrow = rowB + wv * 16 + (lane >> 4) * 8 + r;
                float v = acc[r] + b;
                if (relu) v = fmaxf(v, 0.f);
                if (out_bf16) ((__bf16*)Cout)[(size_t)mrow * ldc + n] = (__bf16)v;
                else          ((float*)Cout)[(size_t)mrow * ldc + n] = v;
            }
        }
    }
}

// ---------------------------------------------------------------------------
// Weight prep kernels
// ---------------------------------------------------------------------------
__global__ void k_w2bf(const float* __restrict__ W, int N, int K,
                       __bf16* __restrict__ dst, int Kp)
{
    int n = blockIdx.x;                        // 0..Np-1
    for (int k = threadIdx.x; k < Kp; k += blockDim.x)
        dst[(size_t)n * Kp + k] =
            (n < N && k < K) ? (__bf16)W[(size_t)n * K + k] : (__bf16)0.f;
}

__global__ void k_padwhh(const float* __restrict__ Whh, float* __restrict__ dst)
{
    int j = blockIdx.x;                        // 0..2047
    for (int k = threadIdx.x; k < 512; k += blockDim.x)
        dst[(size_t)j * 512 + k] =
            (j < G4 && k < HID) ? Whh[(size_t)j * HID + k] : 0.f;
}

// ---------------------------------------------------------------------------
// LSTM recurrence, one direction per block (blockIdx.x: 0=fwd, 1=bwd).
// Whh pre-padded to [2048][512] f32 -> float4 (b128) weight loads.
// ---------------------------------------------------------------------------
__global__ __launch_bounds__(1024)
void k_lstm_dir(const float* __restrict__ preF, const float* __restrict__ preB,
                const float* __restrict__ WpadF, const float* __restrict__ WpadB,
                float* __restrict__ hsF, float* __restrict__ hsB)
{
    const int rev = blockIdx.x;
    const float* __restrict__ pre  = rev ? preB  : preF;
    const float* __restrict__ Wpad = rev ? WpadB : WpadF;
    float* __restrict__ hs         = rev ? hsB   : hsF;

    __shared__ float sh[512];
    __shared__ float sg[2048];
    const int tid = threadIdx.x;
    if (tid < 512) sh[tid] = 0.f;              // sh[511] stays 0 forever
    float c = 0.f;
    __syncthreads();

    for (int t = 0; t < T_STEPS; ++t) {
        const int tt = rev ? (T_STEPS - 1 - t) : t;
        const float* __restrict__ prow = pre + (size_t)tt * LDP;

        if (t + 1 < T_STEPS) {                 // global_prefetch of next gates
            const int tn = rev ? (tt - 1) : (tt + 1);
            __builtin_prefetch(pre + (size_t)tn * LDP + (tid & 2047), 0, 3);
        }

        #pragma unroll
        for (int rr = 0; rr < 2; ++rr) {
            const int j = tid + rr * 1024;     // 0..2047 (pad rows produce junk, unread)
            const float4* __restrict__ w4 = (const float4*)(Wpad + (size_t)j * 512);
            float a = prow[j];
            #pragma unroll 4
            for (int k4 = 0; k4 < 128; ++k4) {
                float4 w = w4[k4];
                a = fmaf(w.x, sh[4 * k4 + 0], a);
                a = fmaf(w.y, sh[4 * k4 + 1], a);
                a = fmaf(w.z, sh[4 * k4 + 2], a);
                a = fmaf(w.w, sh[4 * k4 + 3], a);
            }
            sg[j] = a;
        }
        __syncthreads();

        if (tid < HID) {
            float ig = 1.f / (1.f + __expf(-sg[tid]));
            float fg = 1.f / (1.f + __expf(-sg[HID + tid]));
            float gg = tanhf(sg[2 * HID + tid]);
            float og = 1.f / (1.f + __expf(-sg[3 * HID + tid]));
            c = fg * c + ig * gg;
            float h = og * tanhf(c);
            sh[tid] = h;
            hs[(size_t)tt * HSTR + tid] = h;
        }
        __syncthreads();
    }
}

// ---------------------------------------------------------------------------
// Glue kernels (bf16 activation producers + f32 heads)
// ---------------------------------------------------------------------------
__global__ void k_build_enc_input(const float* __restrict__ ling,
                                  const float* __restrict__ ac,
                                  __bf16* __restrict__ A)
{
    int t = blockIdx.x;
    for (int j = threadIdx.x; j < LDA; j += blockDim.x) {
        float v = 0.f;
        if (j < 442)      v = ling[t * 442 + j];
        else if (j < 641) v = ac[t * 199 + (j - 442)];
        A[(size_t)t * LDA + j] = (__bf16)v;
    }
}

__global__ void k_concat_h(const float* __restrict__ hf,
                           const float* __restrict__ hb,
                           __bf16* __restrict__ dst, int relu)
{
    int t = blockIdx.x;
    for (int j = threadIdx.x; j < LDA; j += blockDim.x) {
        float v = 0.f;
        if (j < HID)          v = hf[(size_t)t * HSTR + j];
        else if (j < 2 * HID) v = hb[(size_t)t * HSTR + (j - HID)];
        if (relu) v = fmaxf(v, 0.f);
        dst[(size_t)t * LDA + j] = (__bf16)v;
    }
}

__global__ void k_zerocols_bf(__bf16* __restrict__ buf, int startCol)
{
    int t = blockIdx.x;
    for (int j = startCol + threadIdx.x; j < LDA; j += blockDim.x)
        buf[(size_t)t * LDA + j] = (__bf16)0.f;
}

__global__ void k_gather_h1(const float* __restrict__ outF,
                            const float* __restrict__ outB,
                            const int* __restrict__ mora,
                            float* __restrict__ H1)
{
    int m  = blockIdx.x;                       // 0..511
    int fi = mora[m];
    int bi = (m == 0) ? 0 : (mora[m - 1] + 1);
    for (int j = threadIdx.x; j < LDA; j += blockDim.x) {
        float v = 0.f;
        if (j < HID)          v = outF[(size_t)fi * HSTR + j];
        else if (j < 2 * HID) v = outB[(size_t)bi * HSTR + (j - HID)];
        H1[(size_t)m * LDA + j] = fmaxf(v, 0.f);
    }
}

__global__ __launch_bounds__(512)
void k_fc2_quant(const float* __restrict__ H1,
                 const float* __restrict__ W, const float* __restrict__ b,
                 const float* __restrict__ codebook,
                 float* __restrict__ zbuf, float* __restrict__ dout)
{
    int m = threadIdx.x;                       // 512 threads, 1 block
    const float* row = H1 + (size_t)m * LDA;
    float a = b[0];
    #pragma unroll 4
    for (int k = 0; k < 2 * HID; ++k) a = fmaf(row[k], W[k], a);
    float c0 = codebook[0], c1 = codebook[1];
    float d0 = (a - c0) * (a - c0), d1 = (a - c1) * (a - c1);
    float zq = (d1 < d0) ? c1 : c0;            // argmin, first index on tie
    zbuf[m] = zq;
    dout[T_STEPS + m]       = zq;              // z
    dout[T_STEPS + 512 + m] = a;               // z_unq
}

__global__ void k_build_dec_input(const float* __restrict__ ling,
                                  const float* __restrict__ zbuf,
                                  const int* __restrict__ mora,
                                  __bf16* __restrict__ A)
{
    int t = blockIdx.x;
    __shared__ int seg;
    if (threadIdx.x == 0) {                    // searchsorted(mora, t, 'right')
        int lo = 0, hi = 512;
        while (lo < hi) { int mid = (lo + hi) >> 1;
                          if (mora[mid] <= t) lo = mid + 1; else hi = mid; }
        seg = lo;
    }
    __syncthreads();
    float zt = (seg < 512) ? zbuf[seg] : 0.f;
    for (int j = threadIdx.x; j < LDA; j += blockDim.x) {
        float v = 0.f;
        if (j < 442)       v = ling[t * 442 + j];
        else if (j == 442) v = zt;
        A[(size_t)t * LDA + j] = (__bf16)v;
    }
}

__global__ void k_fc3(const __bf16* __restrict__ X,
                      const float* __restrict__ W, const float* __restrict__ b,
                      float* __restrict__ dout)
{
    int t = blockIdx.x * blockDim.x + threadIdx.x;
    if (t < T_STEPS) {
        const __bf16* row = X + (size_t)t * LDA;
        float a = b[0];
        #pragma unroll 4
        for (int k = 0; k < 2 * HID; ++k) a = fmaf((float)row[k], W[k], a);
        dout[t] = a;
    }
}

// ---------------------------------------------------------------------------
// Host launcher
// ---------------------------------------------------------------------------
extern "C" void kernel_launch(void* const* d_in, const int* in_sizes, int n_in,
                              void* d_out, int out_size, void* d_ws, size_t ws_size,
                              hipStream_t stream)
{
    const float* ling  = (const float*)d_in[0];
    const float* ac    = (const float*)d_in[1];
    const int*   mora  = (const int*)d_in[2];
    const float* fc11W = (const float*)d_in[4];
    const float* fc11b = (const float*)d_in[5];
    const float* fc2W  = (const float*)d_in[18];
    const float* fc2b  = (const float*)d_in[19];
    const float* cbook = (const float*)d_in[20];
    const float* fc12W = (const float*)d_in[21];
    const float* fc12b = (const float*)d_in[22];
    const float* fc3W  = (const float*)d_in[35];
    const float* fc3b  = (const float*)d_in[36];
    float* dout = (float*)d_out;

    // ---- workspace layout (256B aligned slabs) ----
    size_t off = 0;
    auto carve = [&](size_t bytes) -> void* {
        void* p = (char*)d_ws + off;
        off = (off + bytes + 255) & ~(size_t)255;
        return p;
    };
    __bf16* Abf = (__bf16*)carve((size_t)T_STEPS * LDA * 2);
    __bf16* Bbf = (__bf16*)carve((size_t)T_STEPS * LDA * 2);
    float*  PF  = (float*) carve((size_t)T_STEPS * LDP * 4);
    float*  PB  = (float*) carve((size_t)T_STEPS * LDP * 4);
    float*  HF  = (float*) carve((size_t)T_STEPS * HSTR * 4);
    float*  HB  = (float*) carve((size_t)T_STEPS * HSTR * 4);
    float*  H1  = (float*) carve((size_t)512 * LDA * 4);
    float*  ZB  = (float*) carve(512 * 4);
    float*  WPF = (float*) carve((size_t)G4P * 512 * 4);
    float*  WPB = (float*) carve((size_t)G4P * 512 * 4);
    __bf16* wfc11 = (__bf16*)carve((size_t)704  * 704  * 2);   // Kp=704 (64-mult)
    __bf16* wie0f = (__bf16*)carve((size_t)2048 * 704  * 2);
    __bf16* wie0b = (__bf16*)carve((size_t)2048 * 704  * 2);
    __bf16* wie1f = (__bf16*)carve((size_t)2048 * 1024 * 2);
    __bf16* wie1b = (__bf16*)carve((size_t)2048 * 1024 * 2);
    __bf16* wfc12 = (__bf16*)carve((size_t)448  * 448  * 2);
    __bf16* wid0f = (__bf16*)carve((size_t)2048 * 448  * 2);
    __bf16* wid0b = (__bf16*)carve((size_t)2048 * 448  * 2);
    __bf16* wid1f = (__bf16*)carve((size_t)2048 * 1024 * 2);
    __bf16* wid1b = (__bf16*)carve((size_t)2048 * 1024 * 2);

    const dim3 blk128(128), blk256(256);
    auto cvt = [&](const float* W, int N, int K, __bf16* dst, int Np, int Kp) {
        k_w2bf<<<Np, blk256, 0, stream>>>(W, N, K, dst, Kp);
    };
    auto gemm = [&](const __bf16* Ain, const __bf16* W, int Kp, const float* bias,
                    void* C, int ldc, int N, int Np, int relu, int obf) {
        dim3 grid(Np / 64, T_STEPS / 64);
        k_gemm_wmma<<<grid, blk128, 0, stream>>>(Ain, LDA, W, Kp, bias, C, ldc, N, relu, obf);
    };
    auto lstm = [&](const float* whF, const float* whB) {
        k_padwhh<<<G4P, blk256, 0, stream>>>(whF, WPF);
        k_padwhh<<<G4P, blk256, 0, stream>>>(whB, WPB);
        k_lstm_dir<<<2, 1024, 0, stream>>>(PF, PB, WPF, WPB, HF, HB);
    };

    // ---- one-time (per launch) weight conversion to padded bf16 ----
    cvt(fc11W, 641, 641, wfc11, 704, 704);
    cvt((const float*)d_in[6],  G4, 641,  wie0f, 2048, 704);
    cvt((const float*)d_in[9],  G4, 641,  wie0b, 2048, 704);
    cvt((const float*)d_in[12], G4, 1022, wie1f, 2048, 1024);
    cvt((const float*)d_in[15], G4, 1022, wie1b, 2048, 1024);
    cvt(fc12W, 443, 443, wfc12, 448, 448);
    cvt((const float*)d_in[23], G4, 443,  wid0f, 2048, 448);
    cvt((const float*)d_in[26], G4, 443,  wid0b, 2048, 448);
    cvt((const float*)d_in[29], G4, 1022, wid1f, 2048, 1024);
    cvt((const float*)d_in[32], G4, 1022, wid1b, 2048, 1024);

    // ---------------- encoder ----------------
    k_build_enc_input<<<T_STEPS, blk256, 0, stream>>>(ling, ac, Abf);
    k_zerocols_bf<<<T_STEPS, blk128, 0, stream>>>(Bbf, 641);
    gemm(Abf, wfc11, 704, fc11b, Bbf, LDA, 641, 704, 1, 1);              // fc11+ReLU

    gemm(Bbf, wie0f, 704, (const float*)d_in[8],  PF, LDP, G4, 2048, 0, 0);
    gemm(Bbf, wie0b, 704, (const float*)d_in[11], PB, LDP, G4, 2048, 0, 0);
    lstm((const float*)d_in[7], (const float*)d_in[10]);                 // enc L0
    k_concat_h<<<T_STEPS, blk256, 0, stream>>>(HF, HB, Abf, 0);

    gemm(Abf, wie1f, 1024, (const float*)d_in[14], PF, LDP, G4, 2048, 0, 0);
    gemm(Abf, wie1b, 1024, (const float*)d_in[17], PB, LDP, G4, 2048, 0, 0);
    lstm((const float*)d_in[13], (const float*)d_in[16]);                // enc L1

    // ---------------- bottleneck ----------------
    k_gather_h1<<<512, blk256, 0, stream>>>(HF, HB, mora, H1);
    k_fc2_quant<<<1, 512, 0, stream>>>(H1, fc2W, fc2b, cbook, ZB, dout);

    // ---------------- decoder ----------------
    k_build_dec_input<<<T_STEPS, blk256, 0, stream>>>(ling, ZB, mora, Abf);
    k_zerocols_bf<<<T_STEPS, blk128, 0, stream>>>(Bbf, 443);
    gemm(Abf, wfc12, 448, fc12b, Bbf, LDA, 443, 448, 1, 1);              // fc12+ReLU

    gemm(Bbf, wid0f, 448, (const float*)d_in[25], PF, LDP, G4, 2048, 0, 0);
    gemm(Bbf, wid0b, 448, (const float*)d_in[28], PB, LDP, G4, 2048, 0, 0);
    lstm((const float*)d_in[24], (const float*)d_in[27]);                // dec L0
    k_concat_h<<<T_STEPS, blk256, 0, stream>>>(HF, HB, Abf, 0);

    gemm(Abf, wid1f, 1024, (const float*)d_in[31], PF, LDP, G4, 2048, 0, 0);
    gemm(Abf, wid1b, 1024, (const float*)d_in[34], PB, LDP, G4, 2048, 0, 0);
    lstm((const float*)d_in[30], (const float*)d_in[33]);                // dec L1
    k_concat_h<<<T_STEPS, blk256, 0, stream>>>(HF, HB, Bbf, 1);          // h3 = relu(out)

    k_fc3<<<(T_STEPS + 255) / 256, blk256, 0, stream>>>(Bbf, fc3W, fc3b, dout);
}
